// RNN_circular_LowEtAl_bridged_71081708749399
// MI455X (gfx1250) — matrix operand order repaired and checked by reading, not compiled, not run on hardware
//
#include <hip/hip_runtime.h>

// IRNN scan: h_{t+1} = relu(h_t @ Wh^T + x_t @ Wx^T), emit all states.
// B=256, T=512, H=512, I=2. Output [513, 256, 512] f32.
//
// MI455X strategy (wave32 / gfx1250):
//  - 16 WGs x 16 waves; WG owns a 16-row batch tile -> no cross-WG sync.
//  - Wave w owns h-columns [32w, 32w+32) (two 16-wide N tiles).
//  - Wh (bf16 B-fragments) split: kf 0..9 in VGPRs (160 regs, fits the
//    256-VGPR/wave cap at 4 waves/SIMD), kf 10..15 pre-swizzled in LDS
//    ([half][lane][16B] blocks -> conflict-free ds_load_b128 at lane*16).
//  - h_t in LDS (bf16, double buffered, 520-elem padded rows).
//  - v_wmma_f32_16x16x32_bf16, f32 accumulate; epilogue adds x-projection,
//    ReLU, streams f32 out (non-temporal) + bf16 h to next LDS buffer.
//  - One __syncthreads per timestep.
//  - Dynamic LDS (225 KB) partitioned manually: hbuf | xbuf | wlds.

#define B_     256
#define T_     512
#define H_     512
#define I_     2
#define BT     16      // batch rows per workgroup (WMMA M)
#define NWAVE  16      // waves per workgroup
#define NSLICE 32      // h-columns per wave
#define NT     2       // 16-wide N tiles per wave
#define KF     16      // K fragments of 32 (H_/32)
#define KREG   10      // fragments kept in VGPRs (per N tile)
#define KLDS   (KF - KREG)
#define LDK    520     // padded bf16 row stride for h in LDS (16B pad)

// dynamic LDS partition (bytes)
#define HBUF_BYTES   (2 * BT * LDK * 2)            // 33280
#define XBUF_OFF     HBUF_BYTES
#define XBUF_BYTES   (2 * BT * I_ * 4)             // 256
#define WLDS_OFF     ((XBUF_OFF + XBUF_BYTES + 255) & ~255)   // 33536
#define WLDS_BYTES   (NWAVE * NT * KLDS * 1024)    // 196608
#define SMEM_BYTES   (WLDS_OFF + WLDS_BYTES)       // 230144

typedef __attribute__((ext_vector_type(16))) __bf16 v16bf;
typedef __attribute__((ext_vector_type(8)))  __bf16 v8bf;
typedef __attribute__((ext_vector_type(8)))  float  v8f;

extern __shared__ char smem_raw[];

__global__ __launch_bounds__(NWAVE * 32, 1)
void irnn_scan_kernel(const float* __restrict__ x,
                      const float* __restrict__ Wh,
                      const float* __restrict__ Wx,
                      float* __restrict__ out) {
  __bf16 (*hbuf)[BT][LDK] = (__bf16 (*)[BT][LDK])(smem_raw);
  float  (*xbuf)[BT][I_]  = (float  (*)[BT][I_])(smem_raw + XBUF_OFF);
  __bf16* wlds            = (__bf16*)(smem_raw + WLDS_OFF);

  const int tid  = threadIdx.x;
  const int wave = tid >> 5;
  const int lane = tid & 31;
  const int l15  = lane & 15;
  const int hi16 = lane >> 4;      // 0: lanes 0-15, 1: lanes 16-31
  const int moff = hi16 * 8;       // C/D row offset for this half-wave
  const int b0   = blockIdx.x * BT;

  // ---- preload this wave's Wh slice as WMMA B fragments (f32 -> bf16) ----
  // B[k][n] = Wh[n][k]; lane holds 16 consecutive k of row n:
  //   lanes 0-15: k = kf*32 + 0..15, lanes 16-31: k = kf*32 + 16..31.
  v16bf wreg[NT][KREG];
  float wx0[NT], wx1[NT];
#pragma unroll
  for (int nt = 0; nt < NT; ++nt) {
    const int n = wave * NSLICE + nt * 16 + l15;
    wx0[nt] = Wx[n * I_ + 0];
    wx1[nt] = Wx[n * I_ + 1];
#pragma unroll
    for (int kf = 0; kf < KF; ++kf) {
      const int kb = kf * 32 + hi16 * 16;
      const float* wp = Wh + (size_t)n * H_ + kb;
      v16bf wf;
#pragma unroll
      for (int j = 0; j < 16; ++j) wf[j] = (__bf16)wp[j];
      if (kf < KREG) {
        wreg[nt][kf] = wf;
      } else {
        // LDS-resident fragment: block = [2 halves][32 lanes][8 bf16]
        const int fragElem = ((wave * NT + nt) * KLDS + (kf - KREG)) * 512;
        const v8bf lo = __builtin_shufflevector(wf, wf, 0, 1, 2, 3, 4, 5, 6, 7);
        const v8bf hh = __builtin_shufflevector(wf, wf, 8, 9, 10, 11, 12, 13, 14, 15);
        *(v8bf*)(wlds + fragElem + lane * 8)       = lo;
        *(v8bf*)(wlds + fragElem + 256 + lane * 8) = hh;
      }
    }
  }

  // ---- h0 = one-hot(column 0); also emit out[0] ----
  for (int idx = tid; idx < BT * H_; idx += NWAVE * 32) {
    const int m = idx >> 9;           // H_ == 512
    const int k = idx & (H_ - 1);
    const float v = (k == 0) ? 1.0f : 0.0f;
    hbuf[0][m][k] = (__bf16)v;
    __builtin_nontemporal_store(v, out + (size_t)(b0 + m) * H_ + k);
  }
  // stage x tile for t = 0
  if (tid < BT * I_) {
    const int b = tid >> 1, i = tid & 1;
    xbuf[0][b][i] = x[((size_t)(b0 + b) * T_ + 0) * I_ + i];
  }
  __syncthreads();

  for (int t = 0; t < T_; ++t) {
    const int cb = t & 1;
    const int nb = cb ^ 1;

    // pipeline: stage x tile for t+1 into the other buffer
    if ((t + 1 < T_) && (tid < BT * I_)) {
      const int b = tid >> 1, i = tid & 1;
      xbuf[nb][b][i] = x[((size_t)(b0 + b) * T_ + (t + 1)) * I_ + i];
    }

    // ---- acc = h_t(16xK) * Wh^T slice (Kx16), K = 512 ----
    v8f acc0 = {};
    v8f acc1 = {};
#pragma unroll
    for (int kf = 0; kf < KF; ++kf) {
      // A fragment 16x32 bf16: lane holds M=l15, K = kf*32 + hi16*16 + {0..15}
      const __bf16* hp = &hbuf[cb][l15][kf * 32 + hi16 * 16];
      const v8bf alo = *(const v8bf*)(hp);        // ds_load_b128
      const v8bf ahi = *(const v8bf*)(hp + 8);    // ds_load_b128
      const v16bf a = __builtin_shufflevector(
          alo, ahi, 0, 1, 2, 3, 4, 5, 6, 7, 8, 9, 10, 11, 12, 13, 14, 15);

      v16bf w0, w1;
      if (kf < KREG) {                 // constant under full unroll
        w0 = wreg[0][kf];
        w1 = wreg[1][kf];
      } else {
        const int f0 = ((wave * NT + 0) * KLDS + (kf - KREG)) * 512;
        const int f1 = ((wave * NT + 1) * KLDS + (kf - KREG)) * 512;
        const v8bf w0l = *(const v8bf*)(wlds + f0 + lane * 8);
        const v8bf w0h = *(const v8bf*)(wlds + f0 + 256 + lane * 8);
        const v8bf w1l = *(const v8bf*)(wlds + f1 + lane * 8);
        const v8bf w1h = *(const v8bf*)(wlds + f1 + 256 + lane * 8);
        w0 = __builtin_shufflevector(w0l, w0h,
            0, 1, 2, 3, 4, 5, 6, 7, 8, 9, 10, 11, 12, 13, 14, 15);
        w1 = __builtin_shufflevector(w1l, w1h,
            0, 1, 2, 3, 4, 5, 6, 7, 8, 9, 10, 11, 12, 13, 14, 15);
      }
      acc0 = __builtin_amdgcn_wmma_f32_16x16x32_bf16(
          false, a, false, w0, (short)0, acc0, false, false);
      acc1 = __builtin_amdgcn_wmma_f32_16x16x32_bf16(
          false, a, false, w1, (short)0, acc1, false, false);
    }

    // ---- epilogue: + x_t @ Wx^T, ReLU, write out[t+1] and next h ----
#pragma unroll
    for (int nt = 0; nt < NT; ++nt) {
      const int n = wave * NSLICE + nt * 16 + l15;
      float* outp = out + (((size_t)(t + 1) * B_ + b0) * H_) + n;
#pragma unroll
      for (int r = 0; r < 8; ++r) {
        const int m = moff + r;                       // C/D row for this reg
        const float2 xv = *(const float2*)(&xbuf[cb][m][0]);
        float v = (nt == 0 ? acc0[r] : acc1[r]) + xv.x * wx0[nt] + xv.y * wx1[nt];
        v = fmaxf(v, 0.0f);
        __builtin_nontemporal_store(v, outp + (size_t)m * H_);
        hbuf[nb][m][n] = (__bf16)v;
      }
    }
    __syncthreads();
  }
}

extern "C" void kernel_launch(void* const* d_in, const int* in_sizes, int n_in,
                              void* d_out, int out_size, void* d_ws, size_t ws_size,
                              hipStream_t stream) {
  (void)in_sizes; (void)n_in; (void)d_ws; (void)ws_size; (void)out_size;
  const float* x  = (const float*)d_in[0];   // [256, 512, 2] f32
  const float* Wh = (const float*)d_in[1];   // [512, 512]    f32
  const float* Wx = (const float*)d_in[2];   // [512, 2]      f32
  float* out = (float*)d_out;                // [513, 256, 512] f32
  irnn_scan_kernel<<<dim3(B_ / BT), dim3(NWAVE * 32), SMEM_BYTES, stream>>>(x, Wh, Wx, out);
}
